// SeqCondAttention_10393820856495
// MI455X (gfx1250) — matrix operand order repaired and checked by reading, compiled, not run on hardware
//
#include <hip/hip_runtime.h>

#define KK   12
#define KQn  6
#define HH   64
#define DD   768
#define DIMC 1548   // DIM_CONV
#define DMT  780    // DIM_MEM_TOT
#define OUTC 1536   // OUT_CPLX
#define LL   4096

typedef __attribute__((ext_vector_type(16))) __bf16 v16bf;
typedef __attribute__((ext_vector_type(8)))  float  v8f;
typedef __attribute__((ext_vector_type(4)))  unsigned int u32x4;
typedef __attribute__((ext_vector_type(8)))  int i32x8;
typedef __attribute__((ext_vector_type(4)))  int i32x4;

union FragBF { uint4 u[2]; v16bf v; };

__device__ inline unsigned short f2bf(float f) {
  union { float f; unsigned int u; } v; v.f = f;
  unsigned int r = v.u + 0x7FFFu + ((v.u >> 16) & 1u);
  return (unsigned short)(r >> 16);
}

__device__ inline float silu_f(float x) { return x / (1.f + expf(-x)); }
__device__ inline float softplus_f(float x) { return (x > 20.f) ? x : log1pf(expf(x)); }

// ---------------------------------------------------------------- converts
__global__ void f32_to_bf16_kernel(const float* __restrict__ in,
                                   unsigned short* __restrict__ out, long long n) {
  long long i = (long long)blockIdx.x * blockDim.x + threadIdx.x;
  if (i < n) out[i] = f2bf(in[i]);
}

// W_readout (12,128,384): interleave columns so (val j, gate j) -> (2j, 2j+1)
__global__ void wro_convert_kernel(const float* __restrict__ in,
                                   unsigned short* __restrict__ out) {
  int i = blockIdx.x * 256 + threadIdx.x;
  if (i >= KK * 128 * 384) return;
  int n2 = i % 384, kf = i / 384;
  int j = n2 >> 1;
  int src = kf * 384 + ((n2 & 1) ? (192 + j) : j);
  out[i] = f2bf(in[src]);
}

// ----------------------------------------------------- TDM tile load (B tile)
// 2D tile: tile_dim0 = 64 cols (contiguous), tile_dim1 = 32 rows, element = 2B,
// row stride = ldb elements. Hardware zero-fills reads past tensor_dim0/1.
// LDS receives 32x64 contiguous bf16 (4 KB).
__device__ inline void tdm_load_tile_b(const unsigned short* gptr, unsigned int ldsoff,
                                       int remN, int remK, int ldb)
{
  unsigned long long ga = (unsigned long long)(size_t)gptr;
  u32x4 g0;
  g0.x = 1u;                                       // count = 1 valid descriptor
  g0.y = ldsoff;                                   // lds_addr
  g0.z = (unsigned int)ga;                         // global_addr[31:0]
  g0.w = (unsigned int)((ga >> 32) & 0x1FFFFFFu) | (2u << 30);  // addr[56:32] | type=2
  long long s0 = (long long)ldb;                   // tensor_dim0_stride (elements)
  i32x8 g1;
  g1[0] = 0x10000;                                            // data_size = 1 (2 bytes)
  g1[1] = (int)((remN & 0xFFFF) << 16);                       // tensor_dim0[15:0]
  g1[2] = (int)(((remN >> 16) & 0xFFFF) | ((remK & 0xFFFF) << 16)); // td0 hi | td1 lo
  g1[3] = (int)(((remK >> 16) & 0xFFFF) | (64 << 16));        // td1 hi | tile_dim0=64
  g1[4] = 32;                                                 // tile_dim1=32, tile_dim2=0
  g1[5] = (int)(s0 & 0xFFFFFFFFll);                           // stride0[31:0]
  g1[6] = (int)((s0 >> 32) & 0xFFFF);                         // stride0[47:32] | stride1 lo=0
  g1[7] = 0;
  i32x4 zz = {0, 0, 0, 0};
#if __clang_major__ >= 23
  i32x8 z8 = {0, 0, 0, 0, 0, 0, 0, 0};
  __builtin_amdgcn_tensor_load_to_lds(g0, g1, zz, zz, z8, 0);
#else
  __builtin_amdgcn_tensor_load_to_lds(g0, g1, zz, zz, 0);
#endif
}

// ---------------------------------------------------------------- WMMA GEMM
// C[M,N] = A[M,Kd](bf16,row-major) * B[Kd,N](bf16,row-major), f32 accum.
// Block 128 thr = 4 waves (2x2), block tile 64x64, K-step 32.
// A tile: vectorized global->LDS; B tile: TDM tensor_load_to_lds (zero-fills OOB).
// EPI==0: f32 store to C.  EPI==1: pair-gated bf16 store (readout): even col
// holds val, odd col holds gate; store val*sigmoid(gate) at col/2 + z*192.
template<int EPI>
__global__ void __launch_bounds__(128)
wmma_gemm_kernel(const unsigned short* __restrict__ A,
                 const unsigned short* __restrict__ Bm,
                 float* __restrict__ C, unsigned short* __restrict__ Cbf,
                 int N, int Kd,
                 long long strideA, long long strideB, long long strideC,
                 int lda, int ldb, int ldc)
{
  __shared__ __align__(16) unsigned short As[64 * 40];  // 32 cols + 8 pad
  __shared__ __align__(16) unsigned short Bs[32 * 64];

  const int tid  = threadIdx.x;
  const int lane = tid & 31;
  const int wid  = tid >> 5;
  const int wm   = wid & 1;
  const int wn   = wid >> 1;
  const int half = lane >> 4;
  const int mr   = lane & 15;

  const int z = blockIdx.z;
  const unsigned short* Ag = A + (long long)z * strideA;
  const unsigned short* Bg = Bm + (long long)z * strideB;

  const int rowBase = blockIdx.y * 64;
  const int colBase = blockIdx.x * 64;
  const unsigned int bsOff = (unsigned int)(size_t)(&Bs[0]);

  v8f acc[2][2] = {};

  for (int k0 = 0; k0 < Kd; k0 += 32) {
    // A tile 64x32 (rows guaranteed in range, 16B-aligned vector loads)
    for (int i = tid; i < 256; i += 128) {
      int r = i >> 2;
      int cv = (i & 3) << 3;
      const unsigned short* ap = Ag + (size_t)(rowBase + r) * lda + k0 + cv;
      uint4 val = *(const uint4*)ap;
      *(uint4*)(&As[r * 40 + cv]) = val;
      if (k0 + 32 < Kd) __builtin_prefetch(ap + 32, 0, 1);  // next K-step A rows
    }
    // B tile 32x64 via Tensor Data Mover (one issue per workgroup, wave 0)
    if (wid == 0) {
      tdm_load_tile_b(Bg + (size_t)k0 * ldb + colBase, bsOff,
                      N - colBase, Kd - k0, ldb);
      __builtin_amdgcn_s_wait_tensorcnt(0);
    }
    __syncthreads();

    FragBF a[2], b[2];
    for (int tm = 0; tm < 2; ++tm) {
      const unsigned short* ap = &As[(wm * 32 + tm * 16 + mr) * 40 + 8 * half];
      a[tm].u[0] = *(const uint4*)(ap);       // K = 8h .. 8h+7
      a[tm].u[1] = *(const uint4*)(ap + 16);  // K = 16+8h .. 16+8h+7
    }
    for (int tn = 0; tn < 2; ++tn) {
      const unsigned short* bp = &Bs[(8 * half) * 64 + wn * 32 + tn * 16 + mr];
      b[tn].u[0] = make_uint4((unsigned)bp[0] | ((unsigned)bp[64] << 16), (unsigned)bp[128] | ((unsigned)bp[192] << 16), (unsigned)bp[256] | ((unsigned)bp[320] << 16), (unsigned)bp[384] | ((unsigned)bp[448] << 16));       // K = 8h .. 8h+7
      b[tn].u[1] = make_uint4((unsigned)bp[1024] | ((unsigned)bp[1088] << 16), (unsigned)bp[1152] | ((unsigned)bp[1216] << 16), (unsigned)bp[1280] | ((unsigned)bp[1344] << 16), (unsigned)bp[1408] | ((unsigned)bp[1472] << 16));   // K = 16+8h .. 16+8h+7
    }
    for (int tm = 0; tm < 2; ++tm)
      for (int tn = 0; tn < 2; ++tn)
        acc[tm][tn] = __builtin_amdgcn_wmma_f32_16x16x32_bf16(
            false, a[tm].v, false, b[tn].v, (short)0, acc[tm][tn], false, false);
    __syncthreads();
  }

  for (int tm = 0; tm < 2; ++tm)
    for (int tn = 0; tn < 2; ++tn) {
      int gc = colBase + wn * 32 + tn * 16 + mr;
      for (int v = 0; v < 8; ++v) {
        int gr = rowBase + wm * 32 + tm * 16 + v + 8 * half;
        float val = acc[tm][tn][v];
        if (EPI == 0) {
          if (gc < N) C[(long long)z * strideC + (size_t)gr * ldc + gc] = val;
        } else {
          float p = __shfl_xor(val, 1, 32);   // partner column (gate/val)
          if (((lane & 1) == 0) && gc < N) {
            float sg = 1.f / (1.f + expf(-p));
            Cbf[(size_t)gr * ldc + (size_t)z * 192 + (gc >> 1)] = f2bf(val * sg);
          }
        }
      }
    }
}

// ------------------------------------------------- conv + SiLU + features
// One block per (b,l). Writes stack (p_w | re | im) transposed (B,C,L) and q.
__global__ void __launch_bounds__(256)
conv_feat_kernel(const float* __restrict__ zpre, const float* __restrict__ conv_w,
                 const float* __restrict__ theta_raw, const float* __restrict__ pscale,
                 const float* __restrict__ sscale, const float* __restrict__ sbias,
                 const float* __restrict__ decay,
                 float* __restrict__ stackT, float* __restrict__ qbuf)
{
  __shared__ float z[DIMC];
  __shared__ float pw[KK];
  const int row = blockIdx.x;
  const int b = row >> 12, l = row & (LL - 1);
  const int tid = threadIdx.x;

  for (int c = tid; c < DIMC; c += 256) {
    float acc = 0.f;
    #pragma unroll
    for (int t = 0; t < 4; ++t) {
      int lp = l - 3 + t;
      if (lp >= 0) acc += conv_w[t * DIMC + c] * zpre[((size_t)b * LL + lp) * DIMC + c];
    }
    z[c] = silu_f(acc);
  }
  __syncthreads();

  if (tid < KK) {
    int k = tid;
    float a = sscale[k] * z[DD + k] + sbias[k];
    float p = softplus_f(a);
    float slope = softplus_f(decay[k]);
    p *= expf(-slope * (float)(LL - 1 - l));
    p = fminf(fmaxf(p, 1e-4f), 5000.f);
    pw[k] = p;
    stackT[((size_t)b * DIMC + k) * LL + l] = p;
  }
  __syncthreads();

  for (int c = tid; c < DD; c += 256) {
    int k = c >> 6;
    float kv = z[c];
    float ks = kv * pscale[k];
    float th = 0.001f + 2.999f / (1.f + expf(-theta_raw[c]));
    float phi = ks / (1.f + fabsf(ks)) * th;
    float kvw = kv * pw[k];
    stackT[((size_t)b * DIMC + KK + c) * LL + l]  = kvw * cosf(phi);
    stackT[((size_t)b * DIMC + DMT + c) * LL + l] = kvw * sinf(phi);
  }
  for (int c = tid; c < DD; c += 256)
    qbuf[(size_t)row * DD + c] = z[DMT + c];
}

// ------------------------------------------------- wave-per-chain inclusive scan
// Reads (B,C,L) contiguous, writes (B,L,C).
__global__ void scan_kernel(const float* __restrict__ stackT,
                            float* __restrict__ acc, int chains)
{
  int wid = (blockIdx.x * blockDim.x + threadIdx.x) >> 5;
  int lane = threadIdx.x & 31;
  if (wid >= chains) return;
  int b = wid / DIMC, c = wid % DIMC;
  const float* src = stackT + (size_t)wid * LL;
  float carry = 0.f;
  for (int i = 0; i < LL / 32; ++i) {
    float v = src[i * 32 + lane];
    #pragma unroll
    for (int off = 1; off < 32; off <<= 1) {
      float t = __shfl_up(v, off, 32);
      if (lane >= off) v += t;
    }
    v += carry;
    carry = __shfl(v, 31, 32);
    acc[((size_t)b * LL + i * 32 + lane) * DIMC + c] = v;
  }
}

// ------------------------------------------------- combine + gated RMS-norm
__global__ void __launch_bounds__(256)
combine_kernel(const float* __restrict__ acc, const float* __restrict__ qbuf,
               const float* __restrict__ gate, const float* __restrict__ w_int_raw,
               const float* __restrict__ gnw, unsigned short* __restrict__ onb)
{
  __shared__ float inv_den[KK];
  __shared__ float hv[OUTC];
  __shared__ float rbuf[256];
  const int row = blockIdx.x;
  const int tid = threadIdx.x;

  if (tid < KK) inv_den[tid] = 1.f / fmaxf(acc[(size_t)row * DIMC + tid], 1e-4f);
  __syncthreads();

  float local = 0.f;
  for (int c = tid; c < DD; c += 256) {
    int k = c >> 6, h = c & 63, kq = k >> 1;
    float id  = inv_den[k];
    float sre = acc[(size_t)row * DIMC + KK + c] * id;
    float sim = acc[(size_t)row * DIMC + DMT + c] * id;
    float qre = qbuf[(size_t)row * DD + kq * 128 + h * 2];
    float qim = qbuf[(size_t)row * DD + kq * 128 + h * 2 + 1];
    float mre = (sre * qre + sim * qim) * 0.125f;   // H^-0.5
    float mim = (sim * qre - sre * qim) * 0.125f;
    float wi = expf(w_int_raw[c]);
    float w = wi / (wi + 1e-6f);
    int cre = k * 128 + h, cim = k * 128 + 64 + h;
    float hre = (mre * w) * silu_f(gate[(size_t)row * OUTC + cre]);
    float him = (mim * w) * silu_f(gate[(size_t)row * OUTC + cim]);
    hv[cre] = hre; hv[cim] = him;
    local += hre * hre + him * him;
  }
  rbuf[tid] = local;
  __syncthreads();
  for (int s = 128; s > 0; s >>= 1) {
    if (tid < s) rbuf[tid] += rbuf[tid + s];
    __syncthreads();
  }
  float rms = rsqrtf(rbuf[0] / (float)OUTC + 1e-6f);
  for (int c2 = tid; c2 < OUTC; c2 += 256)
    onb[(size_t)row * OUTC + c2] = f2bf(hv[c2] * rms * gnw[c2]);
}

// ---------------------------------------------------------------- launcher
extern "C" void kernel_launch(void* const* d_in, const int* in_sizes, int n_in,
                              void* d_out, int out_size, void* d_ws, size_t ws_size,
                              hipStream_t stream)
{
  const float* x         = (const float*)d_in[0];
  const float* W_in      = (const float*)d_in[1];
  const float* conv_w    = (const float*)d_in[2];
  const float* theta_raw = (const float*)d_in[3];
  const float* w_int_raw = (const float*)d_in[4];
  const float* decay     = (const float*)d_in[5];
  const float* sscale    = (const float*)d_in[6];
  const float* sbias     = (const float*)d_in[7];
  const float* pscale    = (const float*)d_in[8];
  const float* gnw       = (const float*)d_in[9];
  const float* W_ro      = (const float*)d_in[10];
  const float* W_gate    = (const float*)d_in[11];
  const float* W_out     = (const float*)d_in[12];
  float* out = (float*)d_out;

  const int Bb = in_sizes[0] / (LL * DD);
  const int rows = Bb * LL;

  char* ws = (char*)d_ws;
  size_t off = 0;
  auto alloc = [&](size_t bytes) -> void* {
    void* p = ws + off;
    off += (bytes + 255) & ~(size_t)255;
    return p;
  };
  unsigned short* xb   = (unsigned short*)alloc((size_t)rows * DD * 2);
  unsigned short* winb = (unsigned short*)alloc((size_t)DD * DIMC * 2);
  unsigned short* wgb  = (unsigned short*)alloc((size_t)DD * OUTC * 2);
  unsigned short* wrob = (unsigned short*)alloc((size_t)KK * 128 * 384 * 2);
  unsigned short* wob  = (unsigned short*)alloc((size_t)2304 * DD * 2);
  float* zpre   = (float*)alloc((size_t)rows * DIMC * 4);
  float* gateb  = (float*)alloc((size_t)rows * OUTC * 4);
  float* stackT = (float*)alloc((size_t)Bb * DIMC * LL * 4);
  float* accb   = (float*)alloc((size_t)rows * DIMC * 4);
  float* qbuf   = (float*)alloc((size_t)rows * DD * 4);
  unsigned short* onb = (unsigned short*)alloc((size_t)rows * OUTC * 2);
  unsigned short* ybf = (unsigned short*)alloc((size_t)rows * 2304 * 2);
  (void)ws_size; (void)n_in; (void)out_size;

  long long nx = (long long)rows * DD;
  f32_to_bf16_kernel<<<(int)((nx + 255) / 256), 256, 0, stream>>>(x, xb, nx);
  f32_to_bf16_kernel<<<(DD * DIMC + 255) / 256, 256, 0, stream>>>(W_in, winb, (long long)DD * DIMC);
  f32_to_bf16_kernel<<<(DD * OUTC + 255) / 256, 256, 0, stream>>>(W_gate, wgb, (long long)DD * OUTC);
  f32_to_bf16_kernel<<<(2304 * DD + 255) / 256, 256, 0, stream>>>(W_out, wob, (long long)2304 * DD);
  wro_convert_kernel<<<(KK * 128 * 384 + 255) / 256, 256, 0, stream>>>(W_ro, wrob);

  // z_pre = x @ W_in   [rows x 1548]  (TDM zero-fills the N edge)
  wmma_gemm_kernel<0><<<dim3((DIMC + 63) / 64, rows / 64, 1), 128, 0, stream>>>(
      xb, winb, zpre, nullptr, DIMC, DD, 0, 0, 0, DD, DIMC, DIMC);
  // gate = x @ W_gate  [rows x 1536]
  wmma_gemm_kernel<0><<<dim3(OUTC / 64, rows / 64, 1), 128, 0, stream>>>(
      xb, wgb, gateb, nullptr, OUTC, DD, 0, 0, 0, DD, OUTC, OUTC);

  conv_feat_kernel<<<rows, 256, 0, stream>>>(zpre, conv_w, theta_raw, pscale,
                                             sscale, sbias, decay, stackT, qbuf);

  int chains = Bb * DIMC;
  scan_kernel<<<(chains * 32 + 255) / 256, 256, 0, stream>>>(stackT, accb, chains);

  combine_kernel<<<rows, 256, 0, stream>>>(accb, qbuf, gateb, w_int_raw, gnw, onb);

  // y = gated readout, batched over k (grid.z = 12), fused sigmoid-gate epilogue
  wmma_gemm_kernel<1><<<dim3(384 / 64, rows / 64, KK), 128, 0, stream>>>(
      onb, wrob, nullptr, ybf, 384, 128,
      128, (long long)128 * 384, 0, OUTC, 384, 2304);

  // out = y @ W_out    [rows x 768]
  wmma_gemm_kernel<0><<<dim3(DD / 64, rows / 64, 1), 128, 0, stream>>>(
      ybf, wob, out, nullptr, DD, 2304, 0, 0, 0, 2304, DD, DD);
}
